// _ASPPModuleDeformable_44452911514051
// MI455X (gfx1250) — matrix operand order repaired and checked by reading, not compiled, not run on hardware
//
#include <hip/hip_runtime.h>

typedef __attribute__((ext_vector_type(16))) _Float16 v16h;
typedef __attribute__((ext_vector_type(8)))  float    v8f;

#define CIN 256
#define COUT 256
#define K9 9
#define CK 2304         // CIN * K9
#define M_TOT 16384     // B * H * W

// ---------------------------------------------------------------------------
// Weight packing with PERMUTED reduction order: ckp = k*256 + c.
// Every 32-wide WMMA K-chunk then has uniform k (scalar) and contiguous c.
// Layout [n][ckp] makes each lane's B-fragment one aligned 32B load.
// ---------------------------------------------------------------------------
__global__ __launch_bounds__(256) void pack_w1(const float* __restrict__ w_off,
                                               const float* __restrict__ w_mod,
                                               _Float16* __restrict__ Bg1) {
    int t = blockIdx.x * 256 + threadIdx.x;
    if (t >= 32 * CK) return;
    int n   = t / CK;
    int ckp = t - n * CK;
    int k   = ckp >> 8;
    int c   = ckp & 255;
    int src = c * K9 + k;                    // original (c,k) index
    float v = 0.f;
    if (n < 18)      v = w_off[n * CK + src];
    else if (n < 27) v = w_mod[(n - 18) * CK + src];
    Bg1[t] = (_Float16)v;
}

__global__ __launch_bounds__(256) void pack_w2(const float* __restrict__ w_reg,
                                               _Float16* __restrict__ Bg2) {
    int t = blockIdx.x * 256 + threadIdx.x;
    if (t >= COUT * CK) return;
    int n   = t / CK;
    int ckp = t - n * CK;
    int k   = ckp >> 8;
    int c   = ckp & 255;
    Bg2[t] = (_Float16)w_reg[n * CK + c * K9 + k];
}

// ---------------------------------------------------------------------------
// Offset/mask conv as WMMA GEMM: [16384 x 2304] x [2304 x 32].
// k uniform per outer iteration -> branchless im2col, unconditional loads.
// ---------------------------------------------------------------------------
__global__ __launch_bounds__(256) void conv_offmask_wmma(
    const float* __restrict__ x, const _Float16* __restrict__ Bg1,
    float* __restrict__ raw) {
    int tid  = threadIdx.x;
    int lane = tid & 31;
    int wv   = tid >> 5;
    int blk  = blockIdx.x;
    int mrow0 = (wv & 3) << 4;
    int n0    = (wv >> 2) << 4;
    int lrow  = lane & 15;
    int r  = mrow0 + lrow;
    int m  = blk * 64 + r;
    int b  = m >> 12;
    int hw = m & 4095;
    int h  = hw >> 6;
    int w  = hw & 63;
    const float* xb = x + ((size_t)b << 20);     // b * 256 * 4096
    int k1 = (lane < 16) ? 0 : 16;               // A-frag c base (contiguous, matches B)
    int kb = (lane < 16) ? 0 : 16;               // B-frag K base
    const _Float16* bptr = Bg1 + (size_t)(n0 + lrow) * CK + kb;

    v8f acc = {};
    for (int k = 0; k < K9; ++k) {
        int ky = (k * 11) >> 5;                  // k/3 for k in [0,8]
        int kx = k - ky * 3;
        int y  = h + ky - 1;
        int xx = w + kx - 1;
        float fmask = (((unsigned)y < 64u) && ((unsigned)xx < 64u)) ? 1.f : 0.f;
        int yc = y < 0 ? 0 : (y > 63 ? 63 : y);
        int xc = xx < 0 ? 0 : (xx > 63 ? 63 : xx);
        const float* xs = xb + ((yc << 6) + xc); // per-k base; +c<<12 immediate
        for (int cb = 0; cb < 256; cb += 32) {
            v16h a = {};
#pragma unroll
            for (int e = 0; e < 16; ++e) {
                int c = cb + k1 + e;
                a[e] = (_Float16)(xs[c << 12] * fmask);
            }
            v16h bb = *(const v16h*)(bptr + (k << 8) + cb);
            acc = __builtin_amdgcn_wmma_f32_16x16x32_f16(
                false, a, false, bb, (short)0, acc, false, false);
        }
    }
#pragma unroll
    for (int v = 0; v < 8; ++v) {
        int row = mrow0 + v + ((lane < 16) ? 0 : 8);
        int mm  = blk * 64 + row;
        raw[mm * 32 + n0 + lrow] = acc[v];
    }
}

// ---------------------------------------------------------------------------
// Per (m, k): bias + 2*sigmoid, bilinear corner indices (clamped) and
// weights (validity-zeroed, premultiplied by mask). 32B record.
// ---------------------------------------------------------------------------
__global__ __launch_bounds__(256) void build_meta(
    const float* __restrict__ raw, const float* __restrict__ b_off,
    const float* __restrict__ b_mod, int* __restrict__ meta) {
    int t = blockIdx.x * 256 + threadIdx.x;
    if (t >= M_TOT * K9) return;
    int m = t / K9;
    int k = t - m * K9;

    float dy = raw[m * 32 + 2 * k]     + b_off[2 * k];
    float dx = raw[m * 32 + 2 * k + 1] + b_off[2 * k + 1];
    float mr = raw[m * 32 + 18 + k]    + b_mod[k];
    float mk = 2.f / (1.f + __expf(-mr));

    int hw = m & 4095;
    int h  = hw >> 6;
    int w  = hw & 63;
    int ky = (k * 11) >> 5;
    int kx = k - ky * 3;
    float py = (float)(h + ky - 1) + dy;
    float px = (float)(w + kx - 1) + dx;
    float y0f = floorf(py), x0f = floorf(px);
    float wy = py - y0f, wx = px - x0f;
    int y0 = (int)y0f, x0 = (int)x0f;

    int4 ii; float4 ww;
    {
        int yy = y0, xx = x0;
        bool v = ((unsigned)yy < 64u) && ((unsigned)xx < 64u);
        int yc = yy < 0 ? 0 : (yy > 63 ? 63 : yy);
        int xc = xx < 0 ? 0 : (xx > 63 ? 63 : xx);
        ii.x = (yc << 6) + xc; ww.x = v ? (1.f - wy) * (1.f - wx) * mk : 0.f;
    }
    {
        int yy = y0, xx = x0 + 1;
        bool v = ((unsigned)yy < 64u) && ((unsigned)xx < 64u);
        int yc = yy < 0 ? 0 : (yy > 63 ? 63 : yy);
        int xc = xx < 0 ? 0 : (xx > 63 ? 63 : xx);
        ii.y = (yc << 6) + xc; ww.y = v ? (1.f - wy) * wx * mk : 0.f;
    }
    {
        int yy = y0 + 1, xx = x0;
        bool v = ((unsigned)yy < 64u) && ((unsigned)xx < 64u);
        int yc = yy < 0 ? 0 : (yy > 63 ? 63 : yy);
        int xc = xx < 0 ? 0 : (xx > 63 ? 63 : xx);
        ii.z = (yc << 6) + xc; ww.z = v ? wy * (1.f - wx) * mk : 0.f;
    }
    {
        int yy = y0 + 1, xx = x0 + 1;
        bool v = ((unsigned)yy < 64u) && ((unsigned)xx < 64u);
        int yc = yy < 0 ? 0 : (yy > 63 ? 63 : yy);
        int xc = xx < 0 ? 0 : (xx > 63 ? 63 : xx);
        ii.w = (yc << 6) + xc; ww.w = v ? wy * wx * mk : 0.f;
    }
    ((int4*)meta)[t * 2]       = ii;
    ((float4*)meta)[t * 2 + 1] = ww;
}

// ---------------------------------------------------------------------------
// Fused deform-im2col + WMMA GEMM [16384 x 2304] x [2304 x 256] + BN + ReLU.
// k uniform per outer iteration: one metadata LDS read + 4 hoisted corner
// base pointers per k; all 64 gathers per chunk are immediate-offset loads.
// ---------------------------------------------------------------------------
__global__ __launch_bounds__(256) void deform_gemm_wmma(
    const float* __restrict__ x, const _Float16* __restrict__ Bg2,
    const int* __restrict__ meta, const float* __restrict__ gamma,
    const float* __restrict__ beta, const float* __restrict__ rmean,
    const float* __restrict__ rvar, float* __restrict__ out) {
    __shared__ int4   sI[64 * K9];
    __shared__ float4 sW[64 * K9];

    int tid = threadIdx.x;
    int blk = blockIdx.x;
    for (int rix = tid; rix < 64 * K9; rix += 256) {
        int grec = blk * (64 * K9) + rix;
        sI[rix] = ((const int4*)meta)[grec * 2];
        sW[rix] = ((const float4*)meta)[grec * 2 + 1];
    }
    __syncthreads();

    int lane  = tid & 31;
    int wv    = tid >> 5;
    int mrow0 = (wv & 3) << 4;
    int n0    = (wv >> 2) << 7;                  // 0 or 128
    int lrow  = lane & 15;
    int r     = mrow0 + lrow;
    int m     = blk * 64 + r;
    int b     = m >> 12;
    const float* xb = x + ((size_t)b << 20);
    int k1 = (lane < 16) ? 0 : 16;
    int kb = (lane < 16) ? 0 : 16;
    int rbase = r * K9;
    const _Float16* bptr = Bg2 + (size_t)(n0 + lrow) * CK + kb;

    v8f acc[8];
#pragma unroll
    for (int i = 0; i < 8; ++i) acc[i] = (v8f){};

    for (int k = 0; k < K9; ++k) {
        int4   ii = sI[rbase + k];
        float4 ww = sW[rbase + k];
        const float* p0 = xb + ii.x;
        const float* p1 = xb + ii.y;
        const float* p2 = xb + ii.z;
        const float* p3 = xb + ii.w;
        for (int cb = 0; cb < 256; cb += 32) {
            v16h a = {};
#pragma unroll
            for (int e = 0; e < 16; ++e) {
                int c = cb + k1 + e;
                int o = c << 12;                 // immediate-offset gathers
                float s = ww.x * p0[o] + ww.y * p1[o]
                        + ww.z * p2[o] + ww.w * p3[o];
                a[e] = (_Float16)s;
            }
            int kkp = (k << 8) + cb;
#pragma unroll
            for (int nt = 0; nt < 8; ++nt) {
                v16h bb = *(const v16h*)(bptr + (size_t)nt * (16 * CK) + kkp);
                acc[nt] = __builtin_amdgcn_wmma_f32_16x16x32_f16(
                    false, a, false, bb, (short)0, acc[nt], false, false);
            }
        }
    }

    // Fused inference-BN + ReLU epilogue, vectorized b128 stores
    int mm0 = blk * 64 + mrow0 + ((lane < 16) ? 0 : 8);   // 8 consecutive rows
    int b2  = mm0 >> 12;
    int hw0 = mm0 & 4095;                                  // multiple of 8
    float* ob = out + ((size_t)b2 << 20);                  // b2 * 256 * 4096
#pragma unroll
    for (int nt = 0; nt < 8; ++nt) {
        int o = n0 + (nt << 4) + lrow;
        float inv = gamma[o] * rsqrtf(rvar[o] + 1e-5f);
        float add = beta[o] - rmean[o] * inv;
        float4 lo, hi;
        lo.x = fmaxf(acc[nt][0] * inv + add, 0.f);
        lo.y = fmaxf(acc[nt][1] * inv + add, 0.f);
        lo.z = fmaxf(acc[nt][2] * inv + add, 0.f);
        lo.w = fmaxf(acc[nt][3] * inv + add, 0.f);
        hi.x = fmaxf(acc[nt][4] * inv + add, 0.f);
        hi.y = fmaxf(acc[nt][5] * inv + add, 0.f);
        hi.z = fmaxf(acc[nt][6] * inv + add, 0.f);
        hi.w = fmaxf(acc[nt][7] * inv + add, 0.f);
        float* op = ob + (size_t)o * 4096 + hw0;
        *(float4*)op       = lo;
        *(float4*)(op + 4) = hi;
    }
}

// ---------------------------------------------------------------------------
extern "C" void kernel_launch(void* const* d_in, const int* in_sizes, int n_in,
                              void* d_out, int out_size, void* d_ws, size_t ws_size,
                              hipStream_t stream) {
    (void)in_sizes; (void)n_in; (void)out_size; (void)ws_size;
    const float* x     = (const float*)d_in[0];
    const float* w_off = (const float*)d_in[1];
    const float* b_off = (const float*)d_in[2];
    const float* w_mod = (const float*)d_in[3];
    const float* b_mod = (const float*)d_in[4];
    const float* w_reg = (const float*)d_in[5];
    const float* gamma = (const float*)d_in[6];
    const float* beta  = (const float*)d_in[7];
    const float* rmean = (const float*)d_in[8];
    const float* rvar  = (const float*)d_in[9];

    char* ws = (char*)d_ws;
    _Float16* Bg1 = (_Float16*)(ws + 0);         //   147,456 B  (32 x 2304 f16)
    _Float16* Bg2 = (_Float16*)(ws + 147456);    // 1,179,648 B  (256 x 2304 f16)
    float*    raw = (float*)(ws + 1327104);      // 2,097,152 B  (16384 x 32 f32)
    int*      meta= (int*)(ws + 3424256);        // 4,718,592 B  (147456 x 32B)

    pack_w1<<<dim3((32 * CK + 255) / 256), dim3(256), 0, stream>>>(w_off, w_mod, Bg1);
    pack_w2<<<dim3((COUT * CK + 255) / 256), dim3(256), 0, stream>>>(w_reg, Bg2);
    conv_offmask_wmma<<<dim3(M_TOT / 64), dim3(256), 0, stream>>>(x, Bg1, raw);
    build_meta<<<dim3((M_TOT * K9 + 255) / 256), dim3(256), 0, stream>>>(raw, b_off, b_mod, meta);
    deform_gemm_wmma<<<dim3(M_TOT / 64), dim3(256), 0, stream>>>(
        x, Bg2, meta, gamma, beta, rmean, rvar, (float*)d_out);
}
